// Swinv2Attention_17789754540147
// MI455X (gfx1250) — matrix operand order repaired and checked by reading, not compiled, not run on hardware
//
#include <hip/hip_runtime.h>
#include <hip/hip_bf16.h>
#include <math.h>

// ---------------------------------------------------------------------------
// SwinV2 window attention, bf16-WMMA pipeline for MI455X (gfx1250).
//   B=4096 windows, S=49 tokens, D=384, H=12 heads, d=32 head dim.
// All GEMMs via v_wmma_f32_16x16x32_bf16. Inputs/weights pre-converted to
// bf16 once (weights pre-transposed) so hot-loop LDS tile fills are pure
// 128-bit copies -> async global->LDS path when available.
// ---------------------------------------------------------------------------

#define WSZ 7
#define SEQ 49
#define DIM 384
#define HEADS 12
#define HEAD_DIM 32
#define CPBH 512
#define NWIN 4096
#define TOKENS (NWIN * SEQ) /* 200704 = 1568 * 128 */
#define WELEM (DIM * DIM)   /* 147456 */

typedef __bf16 v16bf __attribute__((ext_vector_type(16)));
typedef float  v8f   __attribute__((ext_vector_type(8)));

union FragU { uint4 q[2]; v16bf v; };

// --- async global->LDS copy (gfx1250 ASYNCcnt path), guarded ---------------
#if defined(__has_builtin)
#  if __has_builtin(__builtin_amdgcn_global_load_async_to_lds_b128)
#    define HAVE_ASYNC_LDS 1
#  endif
#endif
#ifndef HAVE_ASYNC_LDS
#  define HAVE_ASYNC_LDS 0
#endif

#if HAVE_ASYNC_LDS
#  if __has_builtin(__builtin_amdgcn_s_wait_asynccnt)
#    define ASYNC_WAIT() __builtin_amdgcn_s_wait_asynccnt(0)
#  else
#    define ASYNC_WAIT() asm volatile("s_wait_asynccnt 0x0" ::: "memory")
#  endif
typedef int v4i_vs __attribute__((vector_size(16)));
typedef __attribute__((address_space(1))) v4i_vs* as1_v4i;
typedef __attribute__((address_space(3))) v4i_vs* as3_v4i;
__device__ __forceinline__ void async_copy16(const unsigned short* gsrc,
                                             unsigned short* ldst) {
  // generic->AS1 is value-preserving; generic LDS ptr truncates to LDS offset.
  __builtin_amdgcn_global_load_async_to_lds_b128(
      (as1_v4i)(unsigned long long)gsrc,
      (as3_v4i)(unsigned int)(unsigned long long)ldst, 0, 0);
}
#else
#  define ASYNC_WAIT() ((void)0)
#endif

// copy one 64B LDS tile row (4 x b128) from global bf16
__device__ __forceinline__ void tile_row_copy(const unsigned short* gsrc,
                                              unsigned short* ldst) {
#if HAVE_ASYNC_LDS
#pragma unroll
  for (int i = 0; i < 4; ++i) async_copy16(gsrc + 8 * i, ldst + 8 * i);
#else
  const uint4* s4 = (const uint4*)gsrc;
  uint4* d4 = (uint4*)ldst;
#pragma unroll
  for (int i = 0; i < 4; ++i) d4[i] = s4[i];
#endif
}

__device__ __forceinline__ unsigned short f2bf(float f) {
  union { __bf16 b; unsigned short u; } cv;
  cv.b = (__bf16)f; // native v_cvt on gfx1250
  return cv.u;
}
__device__ __forceinline__ float bf2f(unsigned short h) {
  return __uint_as_float(((unsigned int)h) << 16);
}

__device__ __forceinline__ v8f wmma_bf16(const FragU& a, const FragU& b, v8f c) {
  return __builtin_amdgcn_wmma_f32_16x16x32_bf16(
      false, a.v, false, b.v, (short)0, c, false, false);
}

// ---------------------------------------------------------------------------
// Prep kernel: f32 -> bf16 bulk convert (8 elems/thread).
// ---------------------------------------------------------------------------
__global__ void cvt_bf16_kernel(const float* __restrict__ src,
                                unsigned short* __restrict__ dst,
                                long long n8) {
  long long i = blockIdx.x * (long long)blockDim.x + threadIdx.x;
  if (i >= n8) return;
  const float4* s = (const float4*)(src + i * 8);
  float4 a = s[0], b = s[1];
  __align__(16) unsigned short h[8] = {f2bf(a.x), f2bf(a.y), f2bf(a.z), f2bf(a.w),
                                       f2bf(b.x), f2bf(b.y), f2bf(b.z), f2bf(b.w)};
  *(uint4*)(dst + i * 8) = *(const uint4*)h;
}

// Prep kernel: transpose + convert the 4 weight matrices: Wt[n][k] = W[k][n].
__global__ void cvt_w_kernel(const float* __restrict__ Wq,
                             const float* __restrict__ Wk,
                             const float* __restrict__ Wv,
                             const float* __restrict__ Wp,
                             unsigned short* __restrict__ Wt) {
  int which = blockIdx.y;
  const float* W = (which == 0) ? Wq : (which == 1) ? Wk : (which == 2) ? Wv : Wp;
  int i = blockIdx.x * blockDim.x + threadIdx.x; // n*DIM + k
  if (i >= WELEM) return;
  int n = i / DIM, k = i % DIM;
  Wt[(size_t)which * WELEM + i] = f2bf(W[(size_t)k * DIM + n]);
}

// ---------------------------------------------------------------------------
// CPB bias MLP -> bias169[169][12]
// ---------------------------------------------------------------------------
__global__ void cpb_bias_kernel(const float* __restrict__ w0,
                                const float* __restrict__ b0,
                                const float* __restrict__ w1,
                                float* __restrict__ bias169) {
  int t = blockIdx.x * blockDim.x + threadIdx.x;
  for (; t < 169 * HEADS; t += blockDim.x * gridDim.x) {
    int row = t / HEADS, h = t % HEADS;
    float x0 = (float)(row / 13 - 6) * (8.0f / 6.0f);
    float x1 = (float)(row % 13 - 6) * (8.0f / 6.0f);
    float t0 = copysignf(log2f(fabsf(x0) + 1.0f) * (1.0f / 3.0f), x0);
    float t1 = copysignf(log2f(fabsf(x1) + 1.0f) * (1.0f / 3.0f), x1);
    float acc = 0.0f;
    for (int c = 0; c < CPBH; ++c) {
      float hv = t0 * w0[c] + t1 * w0[CPBH + c] + b0[c];
      hv = hv > 0.0f ? hv : 0.0f;
      acc += hv * w1[c * HEADS + h];
    }
    bias169[t] = acc;
  }
}

// Expand to padded per-head tile biasFull[h][64][64] = 16*sigmoid(bias) or
// -1e30 in masked (padded) cells -> attention epilogue is one load + add.
__global__ void bias_expand_kernel(const float* __restrict__ bias169,
                                   float* __restrict__ biasFull) {
  int i = blockIdx.x * blockDim.x + threadIdx.x;
  if (i >= HEADS * 64 * 64) return;
  int h = i >> 12, m = (i >> 6) & 63, n = i & 63;
  float v = -1e30f;
  if (m < SEQ && n < SEQ) {
    int dh = m / WSZ - n / WSZ + (WSZ - 1);
    int dw = m % WSZ - n % WSZ + (WSZ - 1);
    float bv = bias169[(dh * (2 * WSZ - 1) + dw) * HEADS + h];
    v = 16.0f / (1.0f + __expf(-bv));
  }
  biasFull[i] = v;
}

// ---------------------------------------------------------------------------
// GEMM tiling: 128 threads (4 waves), block tile 128x128, K-step 32.
// Wave owns 64x64 = 4x4 WMMA fragments. X tile [m][k], W tile pre-transposed
// [n][k] -> every LDS fill is a contiguous 64B row copy (async b128 path).
// ---------------------------------------------------------------------------
#define GTM 128
#define GTN 128
#define GTK 32
#define TPITCH 40 /* halves: 80B rows keep ds_load_b128 16B-aligned */

// Kernel: QKV projection, bf16 Xb -> bf16 Q/K/V (bias on q,v; none on k).
__global__ __launch_bounds__(128) void qkv_gemm_kernel(
    const unsigned short* __restrict__ Xb, const unsigned short* __restrict__ Wt,
    const float* __restrict__ bq, const float* __restrict__ bv,
    unsigned short* __restrict__ Qws, unsigned short* __restrict__ Kws,
    unsigned short* __restrict__ Vws) {
  __shared__ __align__(16) unsigned short Xl[GTM * TPITCH];
  __shared__ __align__(16) unsigned short Wl[GTN * TPITCH];

  const int tid = threadIdx.x, lane = tid & 31, wave = tid >> 5;
  const int hs = (lane >= 16) ? 1 : 0, lane16 = lane & 15;
  const int M0 = (wave >> 1) * 64, N0 = (wave & 1) * 64;
  const int mb = blockIdx.x, nb = blockIdx.y, which = blockIdx.z;
  const unsigned short* Wsel = Wt + (size_t)which * WELEM;
  const size_t rowBase = (size_t)mb * GTM;

  v8f acc[4][4];
#pragma unroll
  for (int i = 0; i < 4; ++i)
#pragma unroll
    for (int j = 0; j < 4; ++j)
      acc[i][j] = (v8f){0.f, 0.f, 0.f, 0.f, 0.f, 0.f, 0.f, 0.f};

  for (int kk = 0; kk < DIM; kk += GTK) {
    __syncthreads();
    tile_row_copy(Xb + (rowBase + tid) * DIM + kk, &Xl[tid * TPITCH]);
    tile_row_copy(Wsel + ((size_t)nb * GTN + tid) * DIM + kk, &Wl[tid * TPITCH]);
    if (kk + GTK < DIM)
      __builtin_prefetch(Xb + (rowBase + tid) * DIM + kk + GTK, 0, 0);
    ASYNC_WAIT();
    __syncthreads();

    FragU b[4];
#pragma unroll
    for (int nt = 0; nt < 4; ++nt) {
      const uint4* pb = (const uint4*)&Wl[(N0 + nt * 16 + lane16) * TPITCH];
      b[nt].q[0] = pb[hs];
      b[nt].q[1] = pb[2 + hs];
    }
#pragma unroll
    for (int mt = 0; mt < 4; ++mt) {
      FragU a;
      const uint4* pa = (const uint4*)&Xl[(M0 + mt * 16 + lane16) * TPITCH];
      a.q[0] = pa[hs];
      a.q[1] = pa[2 + hs];
#pragma unroll
      for (int nt = 0; nt < 4; ++nt) acc[mt][nt] = wmma_bf16(a, b[nt], acc[mt][nt]);
    }
  }

  unsigned short* Out = (which == 0) ? Qws : ((which == 1) ? Kws : Vws);
  const float* bias = (which == 0) ? bq : ((which == 2) ? bv : nullptr);
#pragma unroll
  for (int mt = 0; mt < 4; ++mt)
#pragma unroll
    for (int nt = 0; nt < 4; ++nt) {
      int col = nb * GTN + N0 + nt * 16 + lane16;
      float badd = bias ? bias[col] : 0.0f;
#pragma unroll
      for (int r = 0; r < 8; ++r) {
        int m = M0 + mt * 16 + r + hs * 8;
        Out[(rowBase + m) * DIM + col] = f2bf(acc[mt][nt][r] + badd);
      }
    }
}

// Kernel: output projection, bf16 O @ Wp^T(pre-transposed bf16) + bp -> f32.
__global__ __launch_bounds__(128) void proj_gemm_kernel(
    const unsigned short* __restrict__ Xb, const unsigned short* __restrict__ Wtp,
    const float* __restrict__ bp, float* __restrict__ Out) {
  __shared__ __align__(16) unsigned short Xl[GTM * TPITCH];
  __shared__ __align__(16) unsigned short Wl[GTN * TPITCH];

  const int tid = threadIdx.x, lane = tid & 31, wave = tid >> 5;
  const int hs = (lane >= 16) ? 1 : 0, lane16 = lane & 15;
  const int M0 = (wave >> 1) * 64, N0 = (wave & 1) * 64;
  const int mb = blockIdx.x, nb = blockIdx.y;
  const size_t rowBase = (size_t)mb * GTM;

  v8f acc[4][4];
#pragma unroll
  for (int i = 0; i < 4; ++i)
#pragma unroll
    for (int j = 0; j < 4; ++j)
      acc[i][j] = (v8f){0.f, 0.f, 0.f, 0.f, 0.f, 0.f, 0.f, 0.f};

  for (int kk = 0; kk < DIM; kk += GTK) {
    __syncthreads();
    tile_row_copy(Xb + (rowBase + tid) * DIM + kk, &Xl[tid * TPITCH]);
    tile_row_copy(Wtp + ((size_t)nb * GTN + tid) * DIM + kk, &Wl[tid * TPITCH]);
    if (kk + GTK < DIM)
      __builtin_prefetch(Xb + (rowBase + tid) * DIM + kk + GTK, 0, 0);
    ASYNC_WAIT();
    __syncthreads();

    FragU b[4];
#pragma unroll
    for (int nt = 0; nt < 4; ++nt) {
      const uint4* pb = (const uint4*)&Wl[(N0 + nt * 16 + lane16) * TPITCH];
      b[nt].q[0] = pb[hs];
      b[nt].q[1] = pb[2 + hs];
    }
#pragma unroll
    for (int mt = 0; mt < 4; ++mt) {
      FragU a;
      const uint4* pa = (const uint4*)&Xl[(M0 + mt * 16 + lane16) * TPITCH];
      a.q[0] = pa[hs];
      a.q[1] = pa[2 + hs];
#pragma unroll
      for (int nt = 0; nt < 4; ++nt) acc[mt][nt] = wmma_bf16(a, b[nt], acc[mt][nt]);
    }
  }

#pragma unroll
  for (int mt = 0; mt < 4; ++mt)
#pragma unroll
    for (int nt = 0; nt < 4; ++nt) {
      int col = nb * GTN + N0 + nt * 16 + lane16;
      float badd = bp[col];
#pragma unroll
      for (int r = 0; r < 8; ++r) {
        int m = M0 + mt * 16 + r + hs * 8;
        Out[(rowBase + m) * DIM + col] = acc[mt][nt][r] + badd;
      }
    }
}

// ---------------------------------------------------------------------------
// Fused attention: one wave32 per (window, head). S padded 49->64.
// q/k normalized lane-locally; exp(logit_scale) folded into q.
// Bias+mask come pre-cooked from biasFull. 16 WMMAs QK^T + 16 WMMAs PV.
// ---------------------------------------------------------------------------
#define QK_PITCH 40
#define VT_PITCH 72
#define S_PITCH 68
#define P_PITCH 72

__global__ __launch_bounds__(32) void attn_kernel(
    const unsigned short* __restrict__ Qws,
    const unsigned short* __restrict__ Kws,
    const unsigned short* __restrict__ Vws,
    const float* __restrict__ biasFull,
    const float* __restrict__ logit_scale,
    unsigned short* __restrict__ Ows) {
  __shared__ __align__(16) unsigned short q_lds[64 * QK_PITCH];
  __shared__ __align__(16) unsigned short k_lds[64 * QK_PITCH];
  __shared__ __align__(16) unsigned short vt_lds[HEAD_DIM * VT_PITCH];
  __shared__ __align__(16) float s_lds[64 * S_PITCH];
  __shared__ __align__(16) unsigned short p_lds[64 * P_PITCH];

  const int h = blockIdx.x, w = blockIdx.y;
  const int lane = threadIdx.x;
  const int hs = (lane >= 16) ? 1 : 0, lane16 = lane & 15;
  const float lscale = __expf(logit_scale[h]);
  const float* bF = biasFull + h * 64 * 64;

  // ---- stage q (normalized*scale), k (normalized), v^T into LDS ----------
#pragma unroll
  for (int rep = 0; rep < 2; ++rep) {
    int t = lane + rep * 32;
    bool valid = t < SEQ;
    float qf[HEAD_DIM], kf[HEAD_DIM], vf[HEAD_DIM];
    if (valid) {
      size_t base = ((size_t)(w * SEQ + t)) * DIM + h * HEAD_DIM;
      const uint4* qp = (const uint4*)(Qws + base);
      const uint4* kp = (const uint4*)(Kws + base);
      const uint4* vp = (const uint4*)(Vws + base);
      __align__(16) uint4 qa[4], ka[4], va[4];
#pragma unroll
      for (int i = 0; i < 4; ++i) { qa[i] = qp[i]; ka[i] = kp[i]; va[i] = vp[i]; }
      const unsigned short* qu = (const unsigned short*)qa;
      const unsigned short* ku = (const unsigned short*)ka;
      const unsigned short* vu = (const unsigned short*)va;
      float ssq = 0.f, ssk = 0.f;
#pragma unroll
      for (int i = 0; i < HEAD_DIM; ++i) {
        qf[i] = bf2f(qu[i]); kf[i] = bf2f(ku[i]); vf[i] = bf2f(vu[i]);
        ssq += qf[i] * qf[i]; ssk += kf[i] * kf[i];
      }
      float rq = lscale / fmaxf(sqrtf(ssq), 1e-6f);
      float rk = 1.0f / fmaxf(sqrtf(ssk), 1e-6f);
#pragma unroll
      for (int i = 0; i < HEAD_DIM; ++i) { qf[i] *= rq; kf[i] *= rk; }
    } else {
#pragma unroll
      for (int i = 0; i < HEAD_DIM; ++i) { qf[i] = 0.f; kf[i] = 0.f; vf[i] = 0.f; }
    }
    __align__(16) unsigned short qh[HEAD_DIM], kh[HEAD_DIM];
#pragma unroll
    for (int i = 0; i < HEAD_DIM; ++i) { qh[i] = f2bf(qf[i]); kh[i] = f2bf(kf[i]); }
    uint4* qd = (uint4*)&q_lds[t * QK_PITCH];
    uint4* kd = (uint4*)&k_lds[t * QK_PITCH];
#pragma unroll
    for (int i = 0; i < 4; ++i) { qd[i] = ((const uint4*)qh)[i]; kd[i] = ((const uint4*)kh)[i]; }
#pragma unroll
    for (int e = 0; e < HEAD_DIM; ++e) vt_lds[e * VT_PITCH + t] = f2bf(vf[e]);
  }
  __syncthreads();

  // ---- S = Q*K^T + biasFull (mask pre-baked) -> s_lds ---------------------
  FragU bk[4];
#pragma unroll
  for (int nt = 0; nt < 4; ++nt) {
    const uint4* pb = (const uint4*)&k_lds[(nt * 16 + lane16) * QK_PITCH];
    bk[nt].q[0] = pb[hs];
    bk[nt].q[1] = pb[2 + hs];
  }
#pragma unroll
  for (int mt = 0; mt < 4; ++mt) {
    FragU a;
    const uint4* pa = (const uint4*)&q_lds[(mt * 16 + lane16) * QK_PITCH];
    a.q[0] = pa[hs];
    a.q[1] = pa[2 + hs];
#pragma unroll
    for (int nt = 0; nt < 4; ++nt) {
      v8f c = (v8f){0.f, 0.f, 0.f, 0.f, 0.f, 0.f, 0.f, 0.f};
      c = wmma_bf16(a, bk[nt], c);
      int n = nt * 16 + lane16;
#pragma unroll
      for (int r = 0; r < 8; ++r) {
        int m = mt * 16 + r + hs * 8;
        s_lds[m * S_PITCH + n] = c[r] + bF[m * 64 + n];
      }
    }
  }
  __syncthreads();

  // ---- softmax rows (2 rows per lane) -> p_lds (bf16) ---------------------
#pragma unroll
  for (int rep = 0; rep < 2; ++rep) {
    int m = lane + rep * 32;
    float rowv[64];
    float mx = -1e30f;
#pragma unroll
    for (int j = 0; j < 64; ++j) { rowv[j] = s_lds[m * S_PITCH + j]; mx = fmaxf(mx, rowv[j]); }
    float sum = 0.f;
#pragma unroll
    for (int j = 0; j < 64; ++j) { float e = __expf(rowv[j] - mx); rowv[j] = e; sum += e; }
    float inv = 1.0f / sum;
    __align__(16) unsigned short ph[64];
#pragma unroll
    for (int j = 0; j < 64; ++j) ph[j] = f2bf(rowv[j] * inv);
    uint4* pd = (uint4*)&p_lds[m * P_PITCH];
#pragma unroll
    for (int i = 0; i < 8; ++i) pd[i] = ((const uint4*)ph)[i];
  }
  __syncthreads();

  // ---- O = P*V -> Ows (bf16) ---------------------------------------------
#pragma unroll
  for (int mt = 0; mt < 4; ++mt) {
#pragma unroll
    for (int nt = 0; nt < 2; ++nt) {
      v8f c = (v8f){0.f, 0.f, 0.f, 0.f, 0.f, 0.f, 0.f, 0.f};
#pragma unroll
      for (int ks = 0; ks < 2; ++ks) {
        FragU a, b;
        const uint4* pa =
            (const uint4*)&p_lds[(mt * 16 + lane16) * P_PITCH + ks * 32];
        a.q[0] = pa[hs];
        a.q[1] = pa[2 + hs];
        const uint4* pb =
            (const uint4*)&vt_lds[(nt * 16 + lane16) * VT_PITCH + ks * 32];
        b.q[0] = pb[hs];
        b.q[1] = pb[2 + hs];
        c = wmma_bf16(a, b, c);
      }
      int n = h * HEAD_DIM + nt * 16 + lane16;
#pragma unroll
      for (int r = 0; r < 8; ++r) {
        int m = mt * 16 + r + hs * 8;
        if (m < SEQ) Ows[((size_t)(w * SEQ + m)) * DIM + n] = f2bf(c[r]);
      }
    }
  }
}

// ---------------------------------------------------------------------------
extern "C" void kernel_launch(void* const* d_in, const int* in_sizes, int n_in,
                              void* d_out, int out_size, void* d_ws,
                              size_t ws_size, hipStream_t stream) {
  const float* hidden = (const float*)d_in[0];
  const float* wq = (const float*)d_in[1];
  const float* bq = (const float*)d_in[2];
  const float* wk = (const float*)d_in[3];
  const float* wv = (const float*)d_in[4];
  const float* bv = (const float*)d_in[5];
  const float* wp = (const float*)d_in[6];
  const float* bp = (const float*)d_in[7];
  const float* cpb_w0 = (const float*)d_in[8];
  const float* cpb_b0 = (const float*)d_in[9];
  const float* cpb_w1 = (const float*)d_in[10];
  const float* logit_scale = (const float*)d_in[11];

  char* ws = (char*)d_ws;
  const size_t mat_bytes = (size_t)TOKENS * DIM * sizeof(unsigned short);
  unsigned short* Qws = (unsigned short*)(ws + 0 * mat_bytes);
  unsigned short* Kws = (unsigned short*)(ws + 1 * mat_bytes);
  unsigned short* Vws = (unsigned short*)(ws + 2 * mat_bytes);
  unsigned short* Ows = (unsigned short*)(ws + 3 * mat_bytes);
  unsigned short* Xb  = (unsigned short*)(ws + 4 * mat_bytes);
  unsigned short* Wt  = (unsigned short*)(ws + 5 * mat_bytes);
  float* bias169  = (float*)(ws + 5 * mat_bytes + 4 * (size_t)WELEM * 2);
  float* biasFull = bias169 + 169 * HEADS;

  const long long n8 = (long long)TOKENS * DIM / 8;
  cvt_bf16_kernel<<<(unsigned)((n8 + 255) / 256), 256, 0, stream>>>(hidden, Xb, n8);
  cvt_w_kernel<<<dim3((WELEM + 255) / 256, 4), 256, 0, stream>>>(wq, wk, wv, wp, Wt);
  cpb_bias_kernel<<<8, 256, 0, stream>>>(cpb_w0, cpb_b0, cpb_w1, bias169);
  bias_expand_kernel<<<(HEADS * 64 * 64 + 255) / 256, 256, 0, stream>>>(bias169,
                                                                        biasFull);

  qkv_gemm_kernel<<<dim3(TOKENS / GTM, DIM / GTN, 3), 128, 0, stream>>>(
      Xb, Wt, bq, bv, Qws, Kws, Vws);

  attn_kernel<<<dim3(HEADS, NWIN), 32, 0, stream>>>(Qws, Kws, Vws, biasFull,
                                                    logit_scale, Ows);

  proj_gemm_kernel<<<dim3(TOKENS / GTM, DIM / GTN), 128, 0, stream>>>(
      Ows, Wt + 3 * (size_t)WELEM, bp, (float*)d_out);
}
